// SelfAttention_19078244729039
// MI455X (gfx1250) — matrix operand order repaired and checked by reading, not compiled, not run on hardware
//
#include <hip/hip_runtime.h>
#include <hip/hip_bf16.h>
#include <math.h>

// ---------------------------------------------------------------------------
// Self-attention for MI455X (gfx1250, wave32, WMMA).
//   N=2, L=2048, EMBED=1024, HEADS=16, HEAD_DIM=64
// All GEMM stages run on v_wmma_f32_16x16x32_f16 (f16 in, f32 accumulate).
// Workspace layout (needs 32 MB):
//   qp  [N,H,L,D] f16   (projected Q)
//   kp  [N,H,L,D] f16   (projected K)
//   vt  [N,H,D,L] f16   (projected V, transposed so PV B-tiles are contiguous)
//   ao  [N,L,EMBED] f16 (attention output, pre-Wout)
// ---------------------------------------------------------------------------

typedef __attribute__((ext_vector_type(16))) _Float16 v16h;
typedef __attribute__((ext_vector_type(8)))  float    v8f;

#define EMBED 1024
#define HEADS 16
#define HDIM  64
#define NB    2
#define SEQL  2048

__device__ __forceinline__ v8f wmma_f16(v16h a, v16h b, v8f c) {
  return __builtin_amdgcn_wmma_f32_16x16x32_f16(
      /*neg_a=*/false, a, /*neg_b=*/false, b,
      /*c_mod=*/(short)0, c, /*reuse_a=*/false, /*reuse_b=*/false);
}

// ---- ISA register layouts (CDNA5 ISA §7.12.2) ----
// A (16x32 f16): lane m = lane&15, half = lane>>4;
//   VGPR pair j holds K = ((j>=4)?16:0) + half*8 + (j&3)*2  and K+1.
// B (32x16 f16): lane n = lane&15, half = lane>>4;
//   VGPR pair j holds K = ((j>=4)?16:0) + half*8 + (j&3)*2  and K+1.
// C/D (16x16 f32): VGPR i -> row = i + 8*half, col = lane&15.

__device__ __forceinline__ v16h load_a_f16(const _Float16* __restrict__ base,
                                           int stride, int lane) {
  const int m = lane & 15, half = lane >> 4;
  v16h a;
#pragma unroll
  for (int j = 0; j < 8; ++j) {
    const int k = ((j >= 4) ? 16 : 0) + half * 8 + (j & 3) * 2;
    const _Float16* p = base + (size_t)m * stride + k;
    a[2 * j]     = p[0];
    a[2 * j + 1] = p[1];
  }
  return a;
}

__device__ __forceinline__ v16h load_a_f32(const float* __restrict__ base,
                                           int stride, int lane) {
  const int m = lane & 15, half = lane >> 4;
  v16h a;
#pragma unroll
  for (int j = 0; j < 8; ++j) {
    const int k = ((j >= 4) ? 16 : 0) + half * 8 + (j & 3) * 2;
    const float* p = base + (size_t)m * stride + k;
    a[2 * j]     = (_Float16)p[0];
    a[2 * j + 1] = (_Float16)p[1];
  }
  return a;
}

__device__ __forceinline__ v16h load_b_f16(const _Float16* __restrict__ base,
                                           int stride, int lane) {
  const int n = lane & 15, half = lane >> 4;
  v16h b;
#pragma unroll
  for (int j = 0; j < 8; ++j) {
    const int k = ((j >= 4) ? 16 : 0) + half * 8 + (j & 3) * 2;
    const _Float16* p = base + (size_t)n * stride + k;
    b[2 * j]     = p[0];
    b[2 * j + 1] = p[1];
  }
  return b;
}

__device__ __forceinline__ v16h load_b_f32(const float* __restrict__ base,
                                           int stride, int lane) {
  const int n = lane & 15, half = lane >> 4;
  v16h b;
#pragma unroll
  for (int j = 0; j < 8; ++j) {
    const int k = ((j >= 4) ? 16 : 0) + half * 8 + (j & 3) * 2;
    const float* p = base + (size_t)n * stride + k;
    b[2 * j]     = (_Float16)p[0];
    b[2 * j + 1] = (_Float16)p[1];
  }
  return b;
}

// ---------------------------------------------------------------------------
// Kernel 1: per-head linear projections.  y = x @ W^T  (rows grouped so each
// 16-row tile is 16 consecutive l for fixed (n,h)).  grid.x = N*H*(L/16),
// grid.y = 3 (q,k,v).  128 threads = 4 waves; wave w owns d-tile w.
// V output stored transposed [N,H,D,L].
// ---------------------------------------------------------------------------
__global__ __launch_bounds__(128) void proj_kernel(
    const float* __restrict__ Xq, const float* __restrict__ Xk,
    const float* __restrict__ Xv,
    const float* __restrict__ Wq, const float* __restrict__ Wk,
    const float* __restrict__ Wv,
    _Float16* __restrict__ qp, _Float16* __restrict__ kp,
    _Float16* __restrict__ vt) {
  const int lane = threadIdx.x & 31;
  const int dt   = threadIdx.x >> 5;      // d-tile 0..3
  const int which = blockIdx.y;           // 0=q 1=k 2=v

  const float* X = (which == 0) ? Xq : (which == 1) ? Xk : Xv;
  const float* W = (which == 0) ? Wq : (which == 1) ? Wk : Wv;

  const int t  = blockIdx.x;
  const int l0 = (t & 127) << 4;
  const int h  = (t >> 7) & 15;
  const int n  = t >> 11;

  // A: 16 rows of the input, head-h columns.
  const float* xbase = X + ((size_t)(n * SEQL + l0)) * EMBED + h * HDIM;
  const v16h a0 = load_a_f32(xbase,      EMBED, lane);
  const v16h a1 = load_a_f32(xbase + 32, EMBED, lane);
  // B[e][d] = W[d][e]: B rows indexed by output dim d.
  const float* wbase = W + (size_t)(dt * 16) * HDIM;
  const v16h b0 = load_b_f32(wbase,      HDIM, lane);
  const v16h b1 = load_b_f32(wbase + 32, HDIM, lane);

  v8f acc = {};
  acc = wmma_f16(a0, b0, acc);
  acc = wmma_f16(a1, b1, acc);

  const int m16 = lane & 15, half = lane >> 4;
  if (which < 2) {
    _Float16* dst = (which == 0) ? qp : kp;
#pragma unroll
    for (int i = 0; i < 8; ++i) {
      const int row = l0 + i + 8 * half;
      const int col = dt * 16 + m16;
      dst[((size_t)(n * HEADS + h) * SEQL + row) * HDIM + col] = (_Float16)acc[i];
    }
  } else {
#pragma unroll
    for (int i = 0; i < 8; ++i) {
      const int row = l0 + i + 8 * half;
      const int col = dt * 16 + m16;
      vt[((size_t)(n * HEADS + h) * HDIM + col) * SEQL + row] = (_Float16)acc[i];
    }
  }
}

// ---------------------------------------------------------------------------
// Kernel 2: flash attention, one wave per (n, h, 16-row q-tile).
// Online softmax over K in 32-column steps:
//   S = Q K^T (4 wmma), mask, row-max/sum via shfl_xor within 16-lane halves,
//   P relaid C->A layout through bank-padded LDS, then 4 PV wmma vs V^T.
// ---------------------------------------------------------------------------
#define PSTR 34   // LDS row stride in halves (68 B -> 17-bank stride, no conflicts)

__global__ __launch_bounds__(32) void attn_kernel(
    const _Float16* __restrict__ qp, const _Float16* __restrict__ kp,
    const _Float16* __restrict__ vt, const int* __restrict__ mask,
    _Float16* __restrict__ ao) {
  __shared__ _Float16 pbuf[16 * PSTR];

  const int lane = threadIdx.x;
  const int t  = blockIdx.x;                 // 0..4095
  const int qt = t & 127;
  const int h  = (t >> 7) & 15;
  const int n  = t >> 11;
  const int q0 = qt << 4;

  const _Float16* Q = qp + ((size_t)(n * HEADS + h) * SEQL + q0) * HDIM;
  const _Float16* K = kp + (size_t)(n * HEADS + h) * SEQL * HDIM;
  const _Float16* V = vt + (size_t)(n * HEADS + h) * HDIM * SEQL;  // [d][l]
  const int*      M = mask + (size_t)n * SEQL * SEQL;

  const int m16 = lane & 15, half = lane >> 4;

  // Q A-tiles (K-dim 64 = two 32-wide chunks), hoisted out of the k loop.
  const v16h qa0 = load_a_f16(Q,      HDIM, lane);
  const v16h qa1 = load_a_f16(Q + 32, HDIM, lane);

  float mst[8], lst[8];
  v8f Ot[4];
#pragma unroll
  for (int i = 0; i < 8; ++i) { mst[i] = -3.0e38f; lst[i] = 0.0f; }
#pragma unroll
  for (int d = 0; d < 4; ++d) Ot[d] = v8f{};

  const float scale = 0.03125f;  // 1/sqrt(EMBED)

#pragma unroll 1
  for (int kb = 0; kb < SEQL; kb += 32) {
    // ---- scores: two 16x16 tiles covering k-cols [kb, kb+32) ----
    v8f s0 = {}, s1 = {};
    {
      const _Float16* Kb0 = K + (size_t)kb * HDIM;
      s0 = wmma_f16(qa0, load_b_f16(Kb0,      HDIM, lane), s0);
      s0 = wmma_f16(qa1, load_b_f16(Kb0 + 32, HDIM, lane), s0);
      const _Float16* Kb1 = K + (size_t)(kb + 16) * HDIM;
      s1 = wmma_f16(qa0, load_b_f16(Kb1,      HDIM, lane), s1);
      s1 = wmma_f16(qa1, load_b_f16(Kb1 + 32, HDIM, lane), s1);
    }

    // ---- scale + mask (reference: where(mask==1, -1e20, e)/32) ----
#pragma unroll
    for (int i = 0; i < 8; ++i) {
      const int q = q0 + i + 8 * half;
      const int k0c = kb + m16;
      const int k1c = kb + 16 + m16;
      float e0 = s0[i] * scale;
      float e1 = s1[i] * scale;
      if (M[(size_t)q * SEQL + k0c] == 1) e0 = -3.125e18f;
      if (M[(size_t)q * SEQL + k1c] == 1) e1 = -3.125e18f;
      s0[i] = e0;
      s1[i] = e1;
    }

    // ---- online softmax statistics (per row, 16-lane half reductions) ----
#pragma unroll
    for (int i = 0; i < 8; ++i) {
      float rm = fmaxf(s0[i], s1[i]);
      rm = fmaxf(rm, __shfl_xor(rm, 1, 32));
      rm = fmaxf(rm, __shfl_xor(rm, 2, 32));
      rm = fmaxf(rm, __shfl_xor(rm, 4, 32));
      rm = fmaxf(rm, __shfl_xor(rm, 8, 32));
      const float mnew = fmaxf(mst[i], rm);
      const float corr = __expf(mst[i] - mnew);
      mst[i] = mnew;
      const float p0 = __expf(s0[i] - mnew);
      const float p1 = __expf(s1[i] - mnew);
      s0[i] = p0;
      s1[i] = p1;
      float rs = p0 + p1;
      rs += __shfl_xor(rs, 1, 32);
      rs += __shfl_xor(rs, 2, 32);
      rs += __shfl_xor(rs, 4, 32);
      rs += __shfl_xor(rs, 8, 32);
      lst[i] = lst[i] * corr + rs;
#pragma unroll
      for (int d = 0; d < 4; ++d) Ot[d][i] *= corr;
    }

    // ---- P: C-layout f32 -> A-layout f16 via LDS transpose ----
#pragma unroll
    for (int i = 0; i < 8; ++i) {
      const int row = i + 8 * half;
      pbuf[row * PSTR + m16]      = (_Float16)s0[i];
      pbuf[row * PSTR + 16 + m16] = (_Float16)s1[i];
    }
    __syncthreads();
    v16h pa;
#pragma unroll
    for (int j = 0; j < 8; ++j) {
      const int k = ((j >= 4) ? 16 : 0) + half * 8 + (j & 3) * 2;
      pa[2 * j]     = pbuf[m16 * PSTR + k];
      pa[2 * j + 1] = pbuf[m16 * PSTR + k + 1];
    }
    __syncthreads();

    // ---- O += P (16x32) x V (32x16 per d-tile), B from V^T rows ----
#pragma unroll
    for (int d = 0; d < 4; ++d) {
      const v16h bv = load_b_f16(V + (size_t)(d * 16) * SEQL + kb, SEQL, lane);
      Ot[d] = wmma_f16(pa, bv, Ot[d]);
    }
  }

  // ---- epilogue: normalize and emit [N,L,EMBED] f16 ----
#pragma unroll
  for (int d = 0; d < 4; ++d) {
#pragma unroll
    for (int i = 0; i < 8; ++i) {
      const int q   = q0 + i + 8 * half;
      const int col = h * HDIM + d * 16 + m16;
      const float o = Ot[d][i] / lst[i];
      ao[(size_t)(n * SEQL + q) * EMBED + col] = (_Float16)o;
    }
  }
}

// ---------------------------------------------------------------------------
// Kernel 3: out = ao @ Wout^T + bout.  4096x1024x1024 GEMM, f32 accumulate.
// grid = (256 m-tiles, 16); 128 threads = 4 waves, wave w -> n-tile 4*by+w.
// ---------------------------------------------------------------------------
__global__ __launch_bounds__(128) void outproj_kernel(
    const _Float16* __restrict__ ao, const float* __restrict__ Wout,
    const float* __restrict__ bout, float* __restrict__ out) {
  const int lane = threadIdx.x & 31;
  const int nt   = blockIdx.y * 4 + (threadIdx.x >> 5);
  const int m0   = blockIdx.x << 4;

  const _Float16* abase = ao + (size_t)m0 * EMBED;
  const float*    wbase = Wout + (size_t)(nt * 16) * EMBED;

  v8f acc = {};
#pragma unroll 4
  for (int kk = 0; kk < EMBED; kk += 32) {
    const v16h a = load_a_f16(abase + kk, EMBED, lane);
    const v16h b = load_b_f32(wbase + kk, EMBED, lane);
    acc = wmma_f16(a, b, acc);
  }

  const int m16 = lane & 15, half = lane >> 4;
  const float bias = bout[nt * 16 + m16];
#pragma unroll
  for (int i = 0; i < 8; ++i) {
    out[(size_t)(m0 + i + 8 * half) * EMBED + nt * 16 + m16] = acc[i] + bias;
  }
}

// ---------------------------------------------------------------------------
extern "C" void kernel_launch(void* const* d_in, const int* in_sizes, int n_in,
                              void* d_out, int out_size, void* d_ws,
                              size_t ws_size, hipStream_t stream) {
  (void)in_sizes; (void)n_in; (void)out_size; (void)ws_size;

  const float* values = (const float*)d_in[0];
  const float* keys   = (const float*)d_in[1];
  const float* query  = (const float*)d_in[2];
  const int*   mask   = (const int*)d_in[3];
  // d_in[4] (inverse_mask) is unused by the reference.
  const float* Wv   = (const float*)d_in[5];
  const float* Wk   = (const float*)d_in[6];
  const float* Wq   = (const float*)d_in[7];
  const float* Wout = (const float*)d_in[8];
  const float* bout = (const float*)d_in[9];
  float* out = (float*)d_out;

  char* ws = (char*)d_ws;
  const size_t SEG = (size_t)NB * HEADS * SEQL * HDIM * sizeof(_Float16);  // 8 MB
  _Float16* qp = (_Float16*)(ws);
  _Float16* kp = (_Float16*)(ws + SEG);
  _Float16* vt = (_Float16*)(ws + 2 * SEG);
  _Float16* ao = (_Float16*)(ws + 3 * SEG);

  // 1) Q/K/V per-head projections (V stored transposed).
  proj_kernel<<<dim3(NB * HEADS * (SEQL / 16), 3), 128, 0, stream>>>(
      query, keys, values, Wq, Wk, Wv, qp, kp, vt);

  // 2) Flash attention: one wave per (n, h, 16-row q-tile).
  attn_kernel<<<NB * HEADS * (SEQL / 16), 32, 0, stream>>>(qp, kp, vt, mask, ao);

  // 3) Output projection + bias.
  outproj_kernel<<<dim3((NB * SEQL) / 16, EMBED / 64), 128, 0, stream>>>(
      ao, Wout, bout, out);
}
